// Attn_31284541784296
// MI455X (gfx1250) — matrix operand order repaired and checked by reading, not compiled, not run on hardware
//
#include <hip/hip_runtime.h>
#include <hip/hip_bf16.h>
#include <math.h>

#define S 8192
#define H 4096
#define A 1024

#define TS 128          // block tile rows (S)
#define TA 128          // block tile cols (A)
#define KC 64           // K chunk staged in LDS (2 WMMA K-steps of 32)
#define NK (H / KC)     // 64 chunks
#define PITCH 72        // LDS row pitch in bf16 elems (144B: 16B aligned, conflict-free)

typedef __attribute__((ext_vector_type(16))) __bf16 v16bf;
typedef __attribute__((ext_vector_type(8)))  __bf16 v8bf;
typedef __attribute__((ext_vector_type(4)))  __bf16 v4bf;
typedef __attribute__((ext_vector_type(8)))  float  v8f;

// CDNA5 hardware tanh (TRANS op). Prefer the builtin (compiler handles the
// TRANS co-execution hazard); fall back to inline asm with explicit V_NOPs.
__device__ __forceinline__ float fast_tanh(float x) {
#if __has_builtin(__builtin_amdgcn_tanhf)
    return __builtin_amdgcn_tanhf(x);
#else
    float r;
    asm volatile("v_tanh_f32 %0, %1\n\tv_nop\n\tv_nop" : "=v"(r) : "v"(x));
    return r;
#endif
}

// ---------------------------------------------------------------------------
// Kernel 0: zero the score accumulator and the output vector.
// ---------------------------------------------------------------------------
__global__ __launch_bounds__(256) void init_zero_kernel(float* __restrict__ scores,
                                                        float* __restrict__ out) {
    int i = blockIdx.x * 256 + threadIdx.x;
    if (i < S) scores[i] = 0.0f;
    if (i < H) out[i]    = 0.0f;
}

// ---------------------------------------------------------------------------
// Kernel 1: cb[a] = hidden . W_w[a,:] + W_b[a] + U_b[a]   (combined bias)
// One wave per output row; lane-strided dot + wave32 shuffle reduction.
// ---------------------------------------------------------------------------
__global__ __launch_bounds__(256) void bias_kernel(const float* __restrict__ hidden,
                                                   const float* __restrict__ Ww,
                                                   const float* __restrict__ Wb,
                                                   const float* __restrict__ Ub,
                                                   float* __restrict__ cb) {
    const int lane = threadIdx.x & 31;
    const int wv   = threadIdx.x >> 5;
    const int a    = blockIdx.x * 8 + wv;
    const float* wr = Ww + (size_t)a * H;
    float s = 0.0f;
    for (int h = lane; h < H; h += 32) s += wr[h] * hidden[h];
    #pragma unroll
    for (int o = 1; o < 32; o <<= 1) s += __shfl_xor(s, o, 32);
    if (lane == 0) cb[a] = s + Wb[a] + Ub[a];
}

// ---------------------------------------------------------------------------
// Kernel 2 (dominant): fused  scores += sum_a V_w[a]*tanh(enc@U_w.T + cb)[.,a]
// bf16 WMMA 16x16x32, f32 accumulate. 128x128 block tile, double-buffered LDS,
// 16 WMMAs per workgroup barrier.
// ---------------------------------------------------------------------------
__global__ __launch_bounds__(256) void attn_scores_kernel(const float* __restrict__ enc,
                                                          const float* __restrict__ Uw,
                                                          const float* __restrict__ cb,
                                                          const float* __restrict__ Vw,
                                                          float* __restrict__ scores) {
    __shared__ __bf16 sE[2][TS * PITCH];   // enc tile (bf16), 2 buffers
    __shared__ __bf16 sU[2][TA * PITCH];   // U_w tile (bf16), 2 buffers

    const int tid  = threadIdx.x;
    const int lane = tid & 31;
    const int wv   = tid >> 5;       // 8 waves
    const int wm   = wv & 3;         // 4 waves along M: 32 rows each
    const int wn   = wv >> 2;        // 2 waves along N: 64 cols each
    const int l16  = lane & 15;
    const int hl   = lane >> 4;      // half-wave id
    const int akb  = hl * 16;        // A-frag K base, contiguous k = 16*hl + s
    const int bkb  = hl * 16;        // B-frag K base, contiguous k = 16*hl + s (same as A)

    const int rowBase = blockIdx.x * TS;
    const int colBase = blockIdx.y * TA;

    v8f acc[2][4] = {};              // 2 M-tiles x 4 N-tiles, f32 C/D

    // Stage one KC-wide chunk of both tiles: f32 global load -> bf16 -> LDS.
    // 128 rows x 16 float4-groups per tile; 8 groups per thread per tile.
    auto stage = [&](int buf, int ks) {
        const int k0 = ks * KC;
        #pragma unroll
        for (int i = 0; i < 8; ++i) {
            const int g  = tid + i * 256;
            const int r  = g >> 4;
            const int kg = (g & 15) << 2;
            float4 e = *(const float4*)(enc + (size_t)(rowBase + r) * H + k0 + kg);
            v4bf pe; pe[0] = (__bf16)e.x; pe[1] = (__bf16)e.y;
                     pe[2] = (__bf16)e.z; pe[3] = (__bf16)e.w;
            *(v4bf*)(&sE[buf][r * PITCH + kg]) = pe;
            float4 u = *(const float4*)(Uw + (size_t)(colBase + r) * H + k0 + kg);
            v4bf pu; pu[0] = (__bf16)u.x; pu[1] = (__bf16)u.y;
                     pu[2] = (__bf16)u.z; pu[3] = (__bf16)u.w;
            *(v4bf*)(&sU[buf][r * PITCH + kg]) = pu;
        }
    };

    stage(0, 0);
    __syncthreads();

    for (int ks = 0; ks < NK; ++ks) {
        const int cur = ks & 1;
        if (ks + 1 < NK) stage(cur ^ 1, ks + 1);   // overlap staging w/ compute

        #pragma unroll
        for (int kk = 0; kk < KC; kk += 32) {
            // A fragments: lane<16 holds K {0..15}, lane>=16 {16..31}
            v16bf afr[2];
            #pragma unroll
            for (int mt = 0; mt < 2; ++mt) {
                const __bf16* p = &sE[cur][(wm * 32 + mt * 16 + l16) * PITCH + kk];
                v8bf lo = *(const v8bf*)(p + akb);
                v8bf hi = *(const v8bf*)(p + akb + 8);
                afr[mt] = __builtin_shufflevector(lo, hi, 0,1,2,3,4,5,6,7,
                                                          8,9,10,11,12,13,14,15);
            }
            // B fragments: lane holds 16 contiguous K of its column, base hl*16
            v16bf bfr[4];
            #pragma unroll
            for (int nt = 0; nt < 4; ++nt) {
                const __bf16* p = &sU[cur][(wn * 64 + nt * 16 + l16) * PITCH + kk];
                v8bf lo = *(const v8bf*)(p + bkb);
                v8bf hi = *(const v8bf*)(p + bkb + 8);
                bfr[nt] = __builtin_shufflevector(lo, hi, 0,1,2,3,4,5,6,7,
                                                          8,9,10,11,12,13,14,15);
            }
            #pragma unroll
            for (int mt = 0; mt < 2; ++mt)
                #pragma unroll
                for (int nt = 0; nt < 4; ++nt)
                    acc[mt][nt] = __builtin_amdgcn_wmma_f32_16x16x32_bf16(
                        false, afr[mt], false, bfr[nt], (short)0, acc[mt][nt],
                        false, false);
        }
        __syncthreads();
    }

    // Epilogue: tanh(acc + cb[col]) * V_w[col]; reduce the 4 in-thread N-tiles,
    // then across the 16 lanes of each half (xor masks stay inside a half in
    // wave32), then one float atomic per row-partial.
    #pragma unroll
    for (int mt = 0; mt < 2; ++mt) {
        float rsum[8] = {};
        #pragma unroll
        for (int nt = 0; nt < 4; ++nt) {
            const int col = colBase + wn * 64 + nt * 16 + l16;
            const float cbv = cb[col];
            const float vwv = Vw[col];
            #pragma unroll
            for (int r = 0; r < 8; ++r)
                rsum[r] += fast_tanh(acc[mt][nt][r] + cbv) * vwv;
        }
        #pragma unroll
        for (int r = 0; r < 8; ++r) {
            float v = rsum[r];
            #pragma unroll
            for (int o = 1; o < 16; o <<= 1) v += __shfl_xor(v, o, 32);
            if (l16 == 0) {
                const int row = rowBase + wm * 32 + mt * 16 + hl * 8 + r;
                atomicAdd(scores + row, v);
            }
        }
    }
}

// ---------------------------------------------------------------------------
// Kernel 3: softmax over S=8192 in a single 1024-thread block.
// (V_b is a uniform shift -> softmax-invariant -> omitted.)
// ---------------------------------------------------------------------------
__global__ __launch_bounds__(1024) void softmax_kernel(const float* __restrict__ scores,
                                                       float* __restrict__ attn) {
    __shared__ float red[32];
    const int tid = threadIdx.x, lane = tid & 31, wv = tid >> 5;

    float v[8];
    #pragma unroll
    for (int i = 0; i < 8; ++i) v[i] = scores[tid + i * 1024];

    float m = v[0];
    #pragma unroll
    for (int i = 1; i < 8; ++i) m = fmaxf(m, v[i]);
    #pragma unroll
    for (int o = 1; o < 32; o <<= 1) m = fmaxf(m, __shfl_xor(m, o, 32));
    if (lane == 0) red[wv] = m;
    __syncthreads();
    if (wv == 0) {
        float x = red[lane];
        #pragma unroll
        for (int o = 1; o < 32; o <<= 1) x = fmaxf(x, __shfl_xor(x, o, 32));
        if (lane == 0) red[0] = x;
    }
    __syncthreads();
    const float M = red[0];
    __syncthreads();

    float s = 0.0f;
    #pragma unroll
    for (int i = 0; i < 8; ++i) { v[i] = expf(v[i] - M); s += v[i]; }
    #pragma unroll
    for (int o = 1; o < 32; o <<= 1) s += __shfl_xor(s, o, 32);
    if (lane == 0) red[wv] = s;
    __syncthreads();
    if (wv == 0) {
        float x = red[lane];
        #pragma unroll
        for (int o = 1; o < 32; o <<= 1) x += __shfl_xor(x, o, 32);
        if (lane == 0) red[0] = x;
    }
    __syncthreads();
    const float inv = 1.0f / red[0];

    #pragma unroll
    for (int i = 0; i < 8; ++i) attn[tid + i * 1024] = v[i] * inv;
}

// ---------------------------------------------------------------------------
// Kernel 4: context[h] = sum_s attn[s] * enc[s,h]  (one coalesced 134MB read)
// ---------------------------------------------------------------------------
#define CSROWS 512
__global__ __launch_bounds__(256) void context_kernel(const float* __restrict__ enc,
                                                      const float* __restrict__ attn,
                                                      float* __restrict__ out) {
    const int col = blockIdx.x * 256 + threadIdx.x;
    const int s0  = blockIdx.y * CSROWS;
    float acc = 0.0f;
    for (int s = 0; s < CSROWS; ++s)
        acc += attn[s0 + s] * enc[(size_t)(s0 + s) * H + col];
    atomicAdd(out + col, acc);
}

// ---------------------------------------------------------------------------
extern "C" void kernel_launch(void* const* d_in, const int* in_sizes, int n_in,
                              void* d_out, int out_size, void* d_ws, size_t ws_size,
                              hipStream_t stream) {
    const float* enc    = (const float*)d_in[0];  // [S,H]
    const float* hidden = (const float*)d_in[1];  // [H]
    const float* Uw     = (const float*)d_in[2];  // [A,H]
    const float* Ub     = (const float*)d_in[3];  // [A]
    const float* Ww     = (const float*)d_in[4];  // [A,H]
    const float* Wb     = (const float*)d_in[5];  // [A]
    const float* Vw     = (const float*)d_in[6];  // [1,A]
    // d_in[7] = V_b: uniform score shift, softmax-invariant -> unused.

    float* out = (float*)d_out;                   // [1,1,H] = H floats

    char*  ws     = (char*)d_ws;                  // needs ~69 KB total
    float* cb     = (float*)ws;                   // A floats
    float* scores = (float*)(ws + A * sizeof(float));            // S floats
    float* attn   = (float*)(ws + (A + S) * sizeof(float));      // S floats

    init_zero_kernel<<<(S + 255) / 256, 256, 0, stream>>>(scores, out);
    bias_kernel<<<A / 8, 256, 0, stream>>>(hidden, Ww, Wb, Ub, cb);
    attn_scores_kernel<<<dim3(S / TS, A / TA), 256, 0, stream>>>(enc, Uw, cb, Vw, scores);
    softmax_kernel<<<1, 1024, 0, stream>>>(scores, attn);
    context_kernel<<<dim3(H / 256, S / CSROWS), 256, 0, stream>>>(enc, attn, out);
}
